// MambaBlock_21560735825986
// MI455X (gfx1250) — matrix operand order repaired and checked by reading, not compiled, not run on hardware
//
#include <hip/hip_runtime.h>
#include <hip/hip_bf16.h>
#include <stdint.h>

typedef __attribute__((ext_vector_type(16))) __bf16 v16bf;
typedef __attribute__((ext_vector_type(8)))  float  v8f;

static constexpr int EMB       = 512;
static constexpr int D_INNER   = 1024;
static constexpr int D_STATE   = 128;
static constexpr int NHEADS    = 16;
static constexpr int HEADDIM   = 64;
static constexpr int CONV_DIM  = D_INNER + 2 * D_STATE;              // 1280
static constexpr int D_IN_PROJ = 2 * D_INNER + 2 * D_STATE + NHEADS; // 2320
static constexpr int SEQ       = 2048;
static constexpr int NBATCH    = 32;            // 4*8
static constexpr int NTOK      = NBATCH * SEQ;  // 65536
static constexpr float EPSF    = 1e-5f;

#define DEV __device__ __forceinline__

DEV unsigned short f2bf(float f) {
    union { float f; uint32_t u; } c; c.f = f;
    uint32_t u = c.u;
    uint32_t r = (u + 0x7FFFu + ((u >> 16) & 1u)) >> 16;   // RNE
    return (unsigned short)r;
}
DEV float sigmoidf_(float x) { return 1.0f / (1.0f + __expf(-x)); }

// ---------------------------------------------------------------------------
// fp32 -> bf16 bulk convert
// ---------------------------------------------------------------------------
__global__ void cvt_bf16_kernel(const float* __restrict__ in,
                                unsigned short* __restrict__ out, long n) {
    long i = (long)blockIdx.x * blockDim.x + threadIdx.x;
    if (i < n) out[i] = f2bf(in[i]);
}

// ---------------------------------------------------------------------------
// LayerNorm over EMB=512, one wave32 per row, output bf16
// ---------------------------------------------------------------------------
__global__ __launch_bounds__(256) void ln_bf16_kernel(
    const float* __restrict__ z, const float* __restrict__ w,
    const float* __restrict__ b, unsigned short* __restrict__ out) {
    int wave = threadIdx.x >> 5, lane = threadIdx.x & 31;
    int row  = blockIdx.x * 8 + wave;
    if (row >= NTOK) return;
    const float* zr = z + (size_t)row * EMB;
    float v[16], s = 0.f, s2 = 0.f;
#pragma unroll
    for (int j = 0; j < 16; ++j) {
        float x = zr[j * 32 + lane];
        v[j] = x; s += x; s2 += x * x;
    }
#pragma unroll
    for (int m = 16; m >= 1; m >>= 1) {
        s  += __shfl_xor(s,  m, 32);
        s2 += __shfl_xor(s2, m, 32);
    }
    float mu  = s  * (1.f / EMB);
    float var = s2 * (1.f / EMB) - mu * mu;
    float rs  = rsqrtf(var + EPSF);
#pragma unroll
    for (int j = 0; j < 16; ++j) {
        int i = j * 32 + lane;
        out[(size_t)row * EMB + i] = f2bf((v[j] - mu) * rs * w[i] + b[i]);
    }
}

// ---------------------------------------------------------------------------
// Generic bf16 WMMA GEMM:  C[M,N] = A[M,K] * B[N,K]^T  (+bias +resid)
// 256 threads = 8 waves; block tile 128x64; wave tile 32x32 (2x2 WMMA 16x16x32)
// K-tile 64, double-buffered LDS fed by GLOBAL_LOAD_ASYNC_TO_LDS_B128.
// Last K-stage is peeled so the steady-state loop is branch-free:
//   issue(next) -> s_wait_asynccnt 6 -> barrier -> 8x WMMA -> barrier
// ---------------------------------------------------------------------------
__global__ __launch_bounds__(256) void wmma_gemm_bf16(
    const unsigned short* __restrict__ A, int lda,
    const unsigned short* __restrict__ Bw, int ldb,
    float* __restrict__ Cf, unsigned short* __restrict__ Cb, int ldc,
    int M, int N, int K,
    const float* __restrict__ bias, const float* __restrict__ resid) {
    constexpr int KT  = 64;   // K per stage
    constexpr int AST = 72;   // padded LDS row stride (halfwords): 144B rows,
                              // 16B-aligned b128 stores, conflict-free banks
    __shared__ unsigned short As[2][128][AST];
    __shared__ unsigned short Bs[2][64][AST];

    const int tid  = threadIdx.x;
    const int lane = tid & 31;
    const int wave = tid >> 5;
    const int row0 = blockIdx.y * 128;
    const int col0 = blockIdx.x * 64;
    const int mBase = (wave & 3) * 32;
    const int nBase = (wave >> 2) * 32;

    v8f acc[2][2] = {};

    // Issue one K-stage of async memory->LDS traffic (6 instrs per wave).
    auto issue = [&](int buf, int k0) {
#pragma unroll
        for (int i = 0; i < 4; ++i) {               // A: 128 x 64 halfwords
            int li = tid + i * 256;
            int r  = li >> 3;
            int kc = (li & 7) * 8;
            int gr = min(row0 + r, M - 1);
            uint64_t ga = (uint64_t)(uintptr_t)(A + (size_t)gr * lda + k0 + kc);
            uint32_t la = (uint32_t)(uintptr_t)&As[buf][r][kc];
            asm volatile("global_load_async_to_lds_b128 %0, %1, off"
                         :: "v"(la), "v"(ga) : "memory");
        }
#pragma unroll
        for (int i = 0; i < 2; ++i) {               // B: 64 x 64 halfwords
            int li = tid + i * 256;
            int r  = li >> 3;
            int kc = (li & 7) * 8;
            int gn = min(col0 + r, N - 1);
            uint64_t ga = (uint64_t)(uintptr_t)(Bw + (size_t)gn * ldb + k0 + kc);
            uint32_t la = (uint32_t)(uintptr_t)&Bs[buf][r][kc];
            asm volatile("global_load_async_to_lds_b128 %0, %1, off"
                         :: "v"(la), "v"(ga) : "memory");
        }
    };

    // Consume one staged K-tile: 2 k-steps x (2x2) WMMA.
    auto compute = [&](int buf) {
        union Fr { v16bf v; uint32_t u[8]; };
        const int fr  = lane & 15;           // M (A) or N (B) index in tile
        const int kB  = (lane >> 4) * 8;     // A low/high-lane K base
        const int kB2 = (lane >> 4) * 16;    // unused; B also reads at kB
#pragma unroll
        for (int ks = 0; ks < KT; ks += 32) {
            Fr a[2], b[2];
#pragma unroll
            for (int mi = 0; mi < 2; ++mi) {
                const uint32_t* ap =
                    (const uint32_t*)&As[buf][mBase + mi * 16 + fr][0];
                int kb = ks + kB;
                a[mi].u[0] = ap[(kb + 0)  >> 1];
                a[mi].u[1] = ap[(kb + 2)  >> 1];
                a[mi].u[2] = ap[(kb + 4)  >> 1];
                a[mi].u[3] = ap[(kb + 6)  >> 1];
                a[mi].u[4] = ap[(kb + 16) >> 1];
                a[mi].u[5] = ap[(kb + 18) >> 1];
                a[mi].u[6] = ap[(kb + 20) >> 1];
                a[mi].u[7] = ap[(kb + 22) >> 1];
            }
#pragma unroll
            for (int ni = 0; ni < 2; ++ni) {
                const uint32_t* bp =
                    (const uint32_t*)&Bs[buf][nBase + ni * 16 + fr][0];
#pragma unroll
                for (int v = 0; v < 8; ++v)
                    b[ni].u[v] = bp[((ks + kB) >> 1) + v + (v >> 2) * 4];
            }
#pragma unroll
            for (int mi = 0; mi < 2; ++mi)
#pragma unroll
                for (int ni = 0; ni < 2; ++ni)
                    acc[mi][ni] = __builtin_amdgcn_wmma_f32_16x16x32_bf16(
                        false, a[mi].v, false, b[ni].v,
                        (short)0, acc[mi][ni], false, false);
        }
    };

    const int nk = K / KT;
    issue(0, 0);
    // steady state: branch-free double-buffer pipeline
    for (int kt = 0; kt < nk - 1; ++kt) {
        const int buf = kt & 1;
        issue(buf ^ 1, (kt + 1) * KT);
        // in-order async completion: <=6 outstanding ==> current stage landed
        asm volatile("s_wait_asynccnt 0x6" ::: "memory");
        __syncthreads();
        compute(buf);
        __syncthreads();   // all waves done reading before buf is overwritten
    }
    // peeled tail stage
    asm volatile("s_wait_asynccnt 0x0" ::: "memory");
    __syncthreads();
    compute((nk - 1) & 1);

    // epilogue: 16x16 f32 C/D layout: VGPR r -> M=r (lanes 0-15), M=r+8
#pragma unroll
    for (int mi = 0; mi < 2; ++mi)
#pragma unroll
        for (int ni = 0; ni < 2; ++ni) {
            int colg  = col0 + nBase + ni * 16 + (lane & 15);
            int rbase = row0 + mBase + mi * 16 + ((lane >> 4) * 8);
            if (colg < N) {
#pragma unroll
                for (int r = 0; r < 8; ++r) {
                    int rowg = rbase + r;
                    if (rowg < M) {
                        float v = acc[mi][ni][r];
                        if (bias)  v += bias[colg];
                        if (resid) v += resid[(size_t)rowg * ldc + colg];
                        if (Cf) Cf[(size_t)rowg * ldc + colg] = v;
                        if (Cb) Cb[(size_t)rowg * ldc + colg] = f2bf(v);
                    }
                }
            }
        }
}

// ---------------------------------------------------------------------------
// dt = softplus(proj_dt + dt_bias);  dA = exp(-exp(A_log)*dt)
// ---------------------------------------------------------------------------
__global__ void dt_kernel(const float* __restrict__ proj,
                          const float* __restrict__ dt_bias,
                          const float* __restrict__ A_log,
                          float* __restrict__ dt, float* __restrict__ dA) {
    int idx = blockIdx.x * blockDim.x + threadIdx.x;
    if (idx >= NTOK * NHEADS) return;
    int h   = idx & (NHEADS - 1);
    int tok = idx >> 4;
    float x = proj[(size_t)tok * D_IN_PROJ + (D_IN_PROJ - NHEADS) + h] + dt_bias[h];
    float d = (x > 20.f) ? x : log1pf(__expf(x));
    dt[idx] = d;
    dA[idx] = __expf(-__expf(A_log[h]) * d);
}

// ---------------------------------------------------------------------------
// Depthwise causal conv (K=4, left zero-pad) + bias + silu
// ---------------------------------------------------------------------------
__global__ void conv_silu_kernel(const float* __restrict__ proj,
                                 const float* __restrict__ cw,
                                 const float* __restrict__ cb,
                                 float* __restrict__ out) {
    long idx = (long)blockIdx.x * blockDim.x + threadIdx.x;
    if (idx >= (long)NTOK * CONV_DIM) return;
    int c    = (int)(idx % CONV_DIM);
    long tok = idx / CONV_DIM;
    int t    = (int)(tok & (SEQ - 1));
    float acc = cb[c];
#pragma unroll
    for (int k = 0; k < 4; ++k) {
        int tt = t - 3 + k;
        if (tt >= 0)
            acc += proj[(size_t)(tok - 3 + k) * D_IN_PROJ + D_INNER + c] * cw[c * 4 + k];
    }
    out[idx] = acc * sigmoidf_(acc);
}

// ---------------------------------------------------------------------------
// Sequential SSM scan. One block per (head, batch); thread p owns state row
// h[p][0..127] in VGPRs. B/C/x broadcast per step through LDS; next step's
// x/B/C segments prefetched (global_prefetch_b8) while the update runs.
// ---------------------------------------------------------------------------
__global__ __launch_bounds__(64) void ssm_scan_kernel(
    const float* __restrict__ xBC, const float* __restrict__ dt,
    const float* __restrict__ dA, float* __restrict__ yraw) {
    const int h = blockIdx.x;     // head
    const int b = blockIdx.y;     // batch
    const int p = threadIdx.x;    // state row (0..63)

    __shared__ float Bsh[D_STATE];
    __shared__ float Csh[D_STATE];
    __shared__ float xsh[HEADDIM];
    __shared__ float sdt, sdA;

    float hreg[D_STATE];
#pragma unroll
    for (int n = 0; n < D_STATE; ++n) hreg[n] = 0.f;

    for (int t = 0; t < SEQ; ++t) {
        const int tok = b * SEQ + t;
        const float* row = xBC + (size_t)tok * CONV_DIM;
        xsh[p]      = row[h * HEADDIM + p];
        Bsh[p]      = row[D_INNER + p];
        Bsh[p + 64] = row[D_INNER + p + 64];
        Csh[p]      = row[D_INNER + D_STATE + p];
        Csh[p + 64] = row[D_INNER + D_STATE + p + 64];
        if (p == 0) {
            sdt = dt[(size_t)tok * NHEADS + h];
            sdA = dA[(size_t)tok * NHEADS + h];
        }
        if (t + 1 < SEQ) {  // hide next step's HBM latency behind the update
            const float* nrow = row + CONV_DIM;
            __builtin_prefetch(nrow + h * HEADDIM + p, 0, 0);
            __builtin_prefetch(nrow + D_INNER + 2 * p, 0, 0);
            __builtin_prefetch(nrow + D_INNER + D_STATE + 2 * p, 0, 0);
        }
        __syncthreads();

        const float a    = sdA;
        const float coef = sdt * xsh[p];
        float acc = 0.f;
#pragma unroll
        for (int n = 0; n < D_STATE; ++n) {
            hreg[n] = fmaf(hreg[n], a, coef * Bsh[n]);
            acc     = fmaf(hreg[n], Csh[n], acc);
        }
        yraw[(size_t)tok * D_INNER + h * HEADDIM + p] = acc;
        __syncthreads();
    }
}

// ---------------------------------------------------------------------------
// y = (yraw + D[h]*x) * silu(zg);  RMSNorm over 1024 * norm_w -> bf16
// ---------------------------------------------------------------------------
__global__ __launch_bounds__(256) void gate_rms_kernel(
    const float* __restrict__ yraw, const float* __restrict__ xBC,
    const float* __restrict__ proj, const float* __restrict__ Dp,
    const float* __restrict__ nw, unsigned short* __restrict__ out) {
    int wave = threadIdx.x >> 5, lane = threadIdx.x & 31;
    int row  = blockIdx.x * 8 + wave;
    if (row >= NTOK) return;
    float yv[32], s2 = 0.f;
#pragma unroll
    for (int j = 0; j < 32; ++j) {
        int i = j * 32 + lane;
        float x  = xBC[(size_t)row * CONV_DIM + i];
        float y  = yraw[(size_t)row * D_INNER + i] + Dp[i >> 6] * x;
        float gz = proj[(size_t)row * D_IN_PROJ + i];
        y *= gz * sigmoidf_(gz);
        yv[j] = y; s2 += y * y;
    }
#pragma unroll
    for (int m = 16; m >= 1; m >>= 1) s2 += __shfl_xor(s2, m, 32);
    float rs = rsqrtf(s2 * (1.f / D_INNER) + EPSF);
#pragma unroll
    for (int j = 0; j < 32; ++j) {
        int i = j * 32 + lane;
        out[(size_t)row * D_INNER + i] = f2bf(yv[j] * rs * nw[i]);
    }
}

// ---------------------------------------------------------------------------
// Weight-normed FF weights: W[n,k] = ff_g[n]*ff_v[n,k]/||ff_v[n,:]|| -> bf16
// ---------------------------------------------------------------------------
__global__ __launch_bounds__(256) void ffw_kernel(
    const float* __restrict__ fv, const float* __restrict__ fg,
    unsigned short* __restrict__ W) {
    int wave = threadIdx.x >> 5, lane = threadIdx.x & 31;
    int row  = blockIdx.x * 8 + wave;
    if (row >= EMB) return;
    const float* vr = fv + (size_t)row * EMB;
    float r[16], s2 = 0.f;
#pragma unroll
    for (int j = 0; j < 16; ++j) {
        float x = vr[j * 32 + lane];
        r[j] = x; s2 += x * x;
    }
#pragma unroll
    for (int m = 16; m >= 1; m >>= 1) s2 += __shfl_xor(s2, m, 32);
    float scale = fg[row] * rsqrtf(s2);
#pragma unroll
    for (int j = 0; j < 16; ++j)
        W[(size_t)row * EMB + j * 32 + lane] = f2bf(r[j] * scale);
}

// ---------------------------------------------------------------------------
extern "C" void kernel_launch(void* const* d_in, const int* in_sizes, int n_in,
                              void* d_out, int out_size, void* d_ws, size_t ws_size,
                              hipStream_t stream) {
    const float* z      = (const float*)d_in[0];
    const float* ln_w   = (const float*)d_in[1];
    const float* ln_b   = (const float*)d_in[2];
    const float* W1     = (const float*)d_in[3];   // (2320, 512)
    const float* conv_w = (const float*)d_in[4];   // (1280, 4)
    const float* conv_b = (const float*)d_in[5];
    const float* dtb    = (const float*)d_in[6];
    const float* A_log  = (const float*)d_in[7];
    const float* Dp     = (const float*)d_in[8];
    const float* norm_w = (const float*)d_in[9];
    const float* W2     = (const float*)d_in[10];  // (512, 1024)
    const float* ff_v   = (const float*)d_in[11];  // (512, 512)
    const float* ff_g   = (const float*)d_in[12];
    const float* ff_b   = (const float*)d_in[13];
    float* out = (float*)d_out;

    char* ws = (char*)d_ws;
    size_t off = 0;
    auto take = [&](size_t bytes) -> void* {
        void* p = ws + off;
        off += (bytes + 255) & ~(size_t)255;
        return p;
    };
    unsigned short* u_bf = (unsigned short*)take((size_t)NTOK * EMB * 2);
    unsigned short* W1b  = (unsigned short*)take((size_t)D_IN_PROJ * EMB * 2);
    float* proj          = (float*)take((size_t)NTOK * D_IN_PROJ * 4);
    float* dtv           = (float*)take((size_t)NTOK * NHEADS * 4);
    float* dAv           = (float*)take((size_t)NTOK * NHEADS * 4);
    float* xact          = (float*)take((size_t)NTOK * CONV_DIM * 4);
    float* yraw          = (float*)take((size_t)NTOK * D_INNER * 4);
    unsigned short* ybf  = (unsigned short*)take((size_t)NTOK * D_INNER * 2);
    unsigned short* W2b  = (unsigned short*)take((size_t)EMB * D_INNER * 2);
    unsigned short* o1bf = (unsigned short*)take((size_t)NTOK * EMB * 2);
    unsigned short* Wffb = (unsigned short*)take((size_t)EMB * EMB * 2);
    (void)ws_size; (void)n_in; (void)in_sizes; (void)out_size;

    // weight conversions
    {
        long n1 = (long)D_IN_PROJ * EMB;
        cvt_bf16_kernel<<<(unsigned)((n1 + 255) / 256), 256, 0, stream>>>(W1, W1b, n1);
        long n2 = (long)EMB * D_INNER;
        cvt_bf16_kernel<<<(unsigned)((n2 + 255) / 256), 256, 0, stream>>>(W2, W2b, n2);
    }
    // LN -> bf16 activations
    ln_bf16_kernel<<<NTOK / 8, 256, 0, stream>>>(z, ln_w, ln_b, u_bf);

    // in_proj GEMM: 65536x512 * (2320x512)^T -> proj fp32
    {
        dim3 grid((D_IN_PROJ + 63) / 64, NTOK / 128);
        wmma_gemm_bf16<<<grid, 256, 0, stream>>>(u_bf, EMB, W1b, EMB,
                                                 proj, nullptr, D_IN_PROJ,
                                                 NTOK, D_IN_PROJ, EMB,
                                                 nullptr, nullptr);
    }
    // dt/dA
    dt_kernel<<<(NTOK * NHEADS + 255) / 256, 256, 0, stream>>>(proj, dtb, A_log, dtv, dAv);
    // conv + silu
    {
        long n = (long)NTOK * CONV_DIM;
        conv_silu_kernel<<<(unsigned)((n + 255) / 256), 256, 0, stream>>>(proj, conv_w, conv_b, xact);
    }
    // SSM scan
    {
        dim3 grid(NHEADS, NBATCH);
        ssm_scan_kernel<<<grid, HEADDIM, 0, stream>>>(xact, dtv, dAv, yraw);
    }
    // gate + rmsnorm -> bf16
    gate_rms_kernel<<<NTOK / 8, 256, 0, stream>>>(yraw, xact, proj, Dp, norm_w, ybf);

    // out_proj GEMM: 65536x1024 * (512x1024)^T -> bf16
    {
        dim3 grid(EMB / 64, NTOK / 128);
        wmma_gemm_bf16<<<grid, 256, 0, stream>>>(ybf, D_INNER, W2b, D_INNER,
                                                 nullptr, o1bf, EMB,
                                                 NTOK, EMB, D_INNER,
                                                 nullptr, nullptr);
    }
    // weight-norm FF weights
    ffw_kernel<<<EMB / 8, 256, 0, stream>>>(ff_v, ff_g, Wffb);
    // FF GEMM + bias + residual(z) -> final output fp32
    {
        dim3 grid(EMB / 64, NTOK / 128);
        wmma_gemm_bf16<<<grid, 256, 0, stream>>>(o1bf, EMB, Wffb, EMB,
                                                 out, nullptr, EMB,
                                                 NTOK, EMB, EMB,
                                                 ff_b, z);
    }
}
